// RNN_22316650070742
// MI455X (gfx1250) — matrix operand order repaired and checked by reading, not compiled, not run on hardware
//
#include <hip/hip_runtime.h>
#include <hip/hip_bf16.h>
#include <math.h>

// ---------------- problem constants ----------------
constexpr int cB  = 32;      // batch
constexpr int cT  = 127;     // timesteps (T-1)
constexpr int cV  = 32000;   // vocab
constexpr int cE  = 300;     // embed dim
constexpr int cH  = 1024;    // hidden
constexpr int cG  = 3 * cH;  // 3072 gates
constexpr int cM  = cB * cT; // 4064 rows (t-major: m = t*32 + b)
constexpr int cKE = 304;     // E padded to mult of 16 (and 4)
constexpr int cMT = cM / 16;     // 254 row tiles
constexpr int cNTO = cV / 16;    // 2000 col tiles of output GEMM
constexpr int cNGO = cNTO / 4;   // 500 groups of 4 col tiles
constexpr int cMG  = (cMT + 7) / 8; // 32 groups of 8 row tiles (one per wave)

typedef __attribute__((ext_vector_type(2)))  float  v2f;
typedef __attribute__((ext_vector_type(8)))  float  v8f;
typedef __attribute__((ext_vector_type(16))) __bf16 v16bf;

// ---------------- fp32 WMMA 16x16 tile (K in steps of 4) ----------------
// Layouts per CDNA5 ISA 7.12.2:
//   A 16x4 f32 : lane m=lane&15 ; a.x=A[m][2*hi], a.y=A[m][2*hi+1]  (hi=lane>>4)
//   B 4x16 f32 : lane n=lane&15 ; b.x=B[2*hi][n], b.y=B[2*hi+1][n]
//   C 16x16    : row = vgpr + 8*hi, col = lane&15
__device__ __forceinline__ v8f wmma_tile_f32(const float* __restrict__ A, int lda,
                                             const float* __restrict__ Bm, int ldb,
                                             int m0, int n0, int K) {
  const int lane = threadIdx.x & 31;
  const int mr   = lane & 15;
  const int hi   = lane >> 4;
  v8f c = {};
  const float* __restrict__ arow = A + (size_t)(m0 + mr) * lda;
  const float* __restrict__ bcol = Bm + (size_t)n0 + mr;
  for (int k = 0; k < K; k += 4) {
    v2f a, b;
    a.x = arow[k + 2 * hi];
    a.y = arow[k + 2 * hi + 1];
    b.x = bcol[(size_t)(k + 2 * hi) * ldb];
    b.y = bcol[(size_t)(k + 2 * hi + 1) * ldb];
    c = __builtin_amdgcn_wmma_f32_16x16x4_f32(false, a, false, b, (short)0, c, false, false);
  }
  return c;
}

__device__ __forceinline__ void store_tile_f32(float* __restrict__ C, int ldc,
                                               int m0, int n0, v8f c,
                                               const float* __restrict__ bias) {
  const int lane = threadIdx.x & 31;
  const int col  = lane & 15;
  const int rb   = (lane >> 4) * 8;
  const float bv = bias ? bias[n0 + col] : 0.0f;
#pragma unroll
  for (int v = 0; v < 8; ++v)
    C[(size_t)(m0 + rb + v) * ldc + n0 + col] = c[v] + bv;
}

__device__ __forceinline__ float sigmf(float x) { return 1.0f / (1.0f + __expf(-x)); }

// ---------------- async-copy helpers (CDNA5 ASYNCcnt path) ----------------
__device__ __forceinline__ void async_g2lds_b128(unsigned lds_off, const void* gaddr) {
  asm volatile("global_load_async_to_lds_b128 %0, %1, off"
               :: "v"(lds_off), "v"((unsigned long long)(uintptr_t)gaddr)
               : "memory");
}
__device__ __forceinline__ void wait_async_le2() {
#if __has_builtin(__builtin_amdgcn_s_wait_asynccnt)
  __builtin_amdgcn_s_wait_asynccnt(2);
#else
  asm volatile("s_wait_asynccnt 0x2" ::: "memory");
#endif
}
__device__ __forceinline__ void wait_async_0() {
#if __has_builtin(__builtin_amdgcn_s_wait_asynccnt)
  __builtin_amdgcn_s_wait_asynccnt(0);
#else
  asm volatile("s_wait_asynccnt 0x0" ::: "memory");
#endif
}

// ---------------- init (ws is poisoned; zero what must be zero) ----------------
__global__ void k_init(unsigned* bar, float* nll, float* cnt, float* h0, float* h1) {
  size_t i = (size_t)blockIdx.x * blockDim.x + threadIdx.x;
  if (i < 8) bar[i] = 0u;
  if (i < cM) { nll[i] = 0.0f; cnt[i] = 0.0f; }
  if (i < (size_t)cB * cH) { h0[i] = 0.0f; h1[i] = 0.0f; }
}

// ---------------- embedding gather (padded to KE) ----------------
__global__ void k_embed(const int* __restrict__ sent, const float* __restrict__ wordvec,
                        float* __restrict__ X) {
  size_t idx = (size_t)blockIdx.x * blockDim.x + threadIdx.x;
  if (idx >= (size_t)cM * cKE) return;
  int m = (int)(idx / cKE), k = (int)(idx % cKE);
  int t = m / cB, b = m % cB;
  float v = 0.0f;
  if (k < cE) {
    int tok = sent[b * 128 + t];               // sent[:, :-1]
    v = wordvec[(size_t)tok * cE + k];
  }
  X[idx] = v;
}

// ---------------- weight transpose to K-major (with K padding) ----------------
__global__ void k_packT(const float* __restrict__ W, float* __restrict__ Wt,
                        int N, int K, int Kp) {
  size_t idx = (size_t)blockIdx.x * blockDim.x + threadIdx.x;
  if (idx >= (size_t)Kp * N) return;
  int k = (int)(idx / N), n = (int)(idx % N);
  Wt[idx] = (k < K) ? W[(size_t)n * K + k] : 0.0f;
}

// ---------------- generic fp32 WMMA GEMM: C = A @ Bm + bias ----------------
__global__ void k_gemm_bias(const float* __restrict__ A, int lda,
                            const float* __restrict__ Bm, int ldb,
                            const float* __restrict__ bias,
                            float* __restrict__ C, int ldc,
                            int Mdim, int Ndim, int K) {
  int wave = (int)((blockIdx.x * (size_t)blockDim.x + threadIdx.x) >> 5);
  int nt_total = Ndim / 16;
  int total = (Mdim / 16) * nt_total;
  if (wave >= total) return;
  int mt = wave / nt_total, nt = wave % nt_total;
  v8f c = wmma_tile_f32(A, lda, Bm, ldb, mt * 16, nt * 16, K);
  store_tile_f32(C, ldc, mt * 16, nt * 16, c, bias);
}

// ---------------- grid-wide barrier for the persistent scan ----------------
__device__ __forceinline__ void gsync(unsigned* cnt, unsigned* gen, unsigned nb) {
  __threadfence();
  __syncthreads();
  if (threadIdx.x == 0) {
    unsigned g = __atomic_load_n(gen, __ATOMIC_ACQUIRE);
    unsigned a = __atomic_fetch_add(cnt, 1u, __ATOMIC_ACQ_REL);
    if (a + 1 == nb) {
      __atomic_store_n(cnt, 0u, __ATOMIC_RELAXED);
      __atomic_fetch_add(gen, 1u, __ATOMIC_ACQ_REL);
    } else {
      while (__atomic_load_n(gen, __ATOMIC_ACQUIRE) == g)
        __builtin_amdgcn_s_sleep(2);
    }
  }
  __syncthreads();
}

// ---------------- persistent 2-layer GRU scan ----------------
__global__ void __launch_bounds__(256) k_scan(
    const float* __restrict__ Gi0,
    const float* __restrict__ Wt_hh0, const float* __restrict__ Wt_ih1,
    const float* __restrict__ Wt_hh1,
    const float* __restrict__ b_hh0, const float* __restrict__ b_ih1,
    const float* __restrict__ b_hh1,
    float* h0, float* h1, float* gh0, float* gi1, float* gh1,
    float* __restrict__ Hs, unsigned* bar) {
  unsigned* bcnt = bar;
  unsigned* bgen = bar + 1;
  const unsigned nb = gridDim.x;
  const int nthreads = (int)(gridDim.x * blockDim.x);
  const int tid = (int)(blockIdx.x * blockDim.x + threadIdx.x);
  const int wave = tid >> 5;
  const int nwaves = nthreads >> 5;
  const int NT = cG / 16;                 // 192 col tiles

  for (int t = 0; t < cT; ++t) {
    // Phase A: gh0 = h0 @ Wt_hh0 + b_hh0   (32 x 3072) -> 384 tiles
    for (int tile = wave; tile < 2 * NT; tile += nwaves) {
      int mt = tile / NT, nt = tile % NT;
      v8f c = wmma_tile_f32(h0, cH, Wt_hh0, cG, mt * 16, nt * 16, cH);
      store_tile_f32(gh0, cG, mt * 16, nt * 16, c, b_hh0);
    }
    gsync(bcnt, bgen, nb);
    // Gate update layer 0
    for (int i = tid; i < cB * cH; i += nthreads) {
      int b = i / cH, j = i % cH;
      const float* gi = Gi0 + ((size_t)t * cB + b) * cG;
      const float* gh = gh0 + (size_t)b * cG;
      float r = sigmf(gi[j] + gh[j]);
      float z = sigmf(gi[cH + j] + gh[cH + j]);
      float n = tanhf(gi[2 * cH + j] + r * gh[2 * cH + j]);
      h0[i] = (1.0f - z) * n + z * h0[i];
    }
    gsync(bcnt, bgen, nb);
    // Phase B: gi1 = h0' @ Wt_ih1 + b_ih1 ; gh1 = h1 @ Wt_hh1 + b_hh1 -> 768 tiles
    for (int tile = wave; tile < 4 * NT; tile += nwaves) {
      int which = tile / (2 * NT);
      int tt = tile % (2 * NT);
      int mt = tt / NT, nt = tt % NT;
      if (which == 0) {
        v8f c = wmma_tile_f32(h0, cH, Wt_ih1, cG, mt * 16, nt * 16, cH);
        store_tile_f32(gi1, cG, mt * 16, nt * 16, c, b_ih1);
      } else {
        v8f c = wmma_tile_f32(h1, cH, Wt_hh1, cG, mt * 16, nt * 16, cH);
        store_tile_f32(gh1, cG, mt * 16, nt * 16, c, b_hh1);
      }
    }
    gsync(bcnt, bgen, nb);
    // Gate update layer 1 + record Hs[t]
    for (int i = tid; i < cB * cH; i += nthreads) {
      int b = i / cH, j = i % cH;
      const float* gi = gi1 + (size_t)b * cG;
      const float* gh = gh1 + (size_t)b * cG;
      float r = sigmf(gi[j] + gh[j]);
      float z = sigmf(gi[cH + j] + gh[cH + j]);
      float n = tanhf(gi[2 * cH + j] + r * gh[2 * cH + j]);
      float hn = (1.0f - z) * n + z * h1[i];
      h1[i] = hn;
      Hs[((size_t)t * cB + b) * cH + j] = hn;
    }
    gsync(bcnt, bgen, nb);
  }
}

// ---------------- split-bf16 panel packing (per-lane WMMA layout) ----------------
// A 16x32 bf16 (ISA 7.12.2): lane<16 -> K {2v+s}=0..7 in v0..3, 16..23 in v4..7;
//                            lane>=16 -> 8..15 and 24..31.  m = lane&15.
__global__ void k_pack_a(const float* __restrict__ Hs, __bf16* __restrict__ Ahi,
                         __bf16* __restrict__ Alo) {
  size_t idx = (size_t)blockIdx.x * blockDim.x + threadIdx.x;
  if (idx >= ((size_t)cMT << 14)) return;          // 254*32*32*16
  int h    = (int)(idx & 15);
  int lane = (int)((idx >> 4) & 31);
  int kb   = (int)((idx >> 9) & 31);
  int mt   = (int)(idx >> 14);
  int hi = lane >> 4, mr = lane & 15;
  int K = (h < 8) ? (h + 8 * hi) : (16 + (h - 8) + 8 * hi);
  float x = Hs[(size_t)(mt * 16 + mr) * cH + kb * 32 + K];
  __bf16 xh = (__bf16)x;
  Ahi[idx] = xh;
  Alo[idx] = (__bf16)(x - (float)xh);
}

// B 32x16 bf16: lanes 0..15 hold K=0..7,16..23; lanes 16..31 hold K=8..15,24..31; n = lane&15.
// Bmat[k][n] = W_out[n][k].
__global__ void k_pack_b(const float* __restrict__ Wout, __bf16* __restrict__ Bhi,
                         __bf16* __restrict__ Blo) {
  size_t idx = (size_t)blockIdx.x * blockDim.x + threadIdx.x;
  if (idx >= ((size_t)cNTO << 14)) return;         // 2000*32*32*16
  int h    = (int)(idx & 15);
  int lane = (int)((idx >> 4) & 31);
  int kb   = (int)((idx >> 9) & 31);
  int nt   = (int)(idx >> 14);
  int hi = lane >> 4, nr = lane & 15;
  int K = (h < 8) ? (h + 8 * hi) : (16 + (h - 8) + 8 * hi);
  float x = Wout[(size_t)(nt * 16 + nr) * cH + kb * 32 + K];
  __bf16 xh = (__bf16)x;
  Bhi[idx] = xh;
  Blo[idx] = (__bf16)(x - (float)xh);
}

// ---------------- output GEMM: logits = Hs @ W_out^T + b_out ----------------
// Block = 8 waves: wave w owns m-tile (mg*8+w); all waves share one group of 4
// n-tiles. B hi/lo panels (8 KB per k-step) are staged into LDS with
// global_load_async_to_lds_b128 (double-buffered on ASYNCcnt), then consumed
// via ds loads. bf16x3 split recovers ~fp32 accuracy with 3 WMMA per tile.
__global__ void __launch_bounds__(256) k_gemm_out(
    const __bf16* __restrict__ Ahi, const __bf16* __restrict__ Alo,
    const __bf16* __restrict__ Bhi, const __bf16* __restrict__ Blo,
    const float* __restrict__ b_out, float* __restrict__ out /* d_out */) {
  __shared__ __align__(16) __bf16 Bbuf[2][4][2][512];   // [buf][ntile][hi/lo][1KB]
  const int tid  = threadIdx.x;
  const int lane = tid & 31;
  const int wv   = tid >> 5;                            // 0..7
  const int mg = blockIdx.x / cNGO;                     // 0..31
  const int ng = blockIdx.x % cNGO;                     // 0..499
  const int mt = mg * 8 + wv;
  const bool valid = (mt < cMT);
  const int mtc = valid ? mt : (cMT - 1);

  // stage B panels for k-block kb into Bbuf[buf]
  const int jC = tid >> 6;                // chunk 0..3 (64 threads each)
  const int oC = (tid & 63) * 16;         // byte offset within 1KB chunk
  const char* BhiB = (const char*)Bhi;
  const char* BloB = (const char*)Blo;
  auto stage = [&](int kb, int buf) {
    size_t gb = ((((size_t)(ng * 4 + jC)) * 32 + kb) * 512) * 2 + oC;
    unsigned lh = (unsigned)(uintptr_t)(&Bbuf[buf][jC][0][0]) + oC;
    unsigned ll = (unsigned)(uintptr_t)(&Bbuf[buf][jC][1][0]) + oC;
    async_g2lds_b128(lh, BhiB + gb);
    async_g2lds_b128(ll, BloB + gb);
  };

  stage(0, 0);
  v8f acc[4] = {{}, {}, {}, {}};
  for (int kb = 0; kb < 32; ++kb) {
    const int cur = kb & 1;
    if (kb + 1 < 32) { stage(kb + 1, cur ^ 1); wait_async_le2(); }
    else             { wait_async_0(); }
    __syncthreads();                      // staged panels visible to all waves

    size_t aoff = (((size_t)mtc * 32 + kb) * 32 + lane) * 16;
    v16bf ah = *(const v16bf*)(Ahi + aoff);
    v16bf al = *(const v16bf*)(Alo + aoff);
    if (kb + 1 < 32) {
      __builtin_prefetch(Ahi + aoff + 512, 0, 1);
      __builtin_prefetch(Alo + aoff + 512, 0, 1);
    }
#pragma unroll
    for (int j = 0; j < 4; ++j) {
      v16bf bh = *(const v16bf*)(&Bbuf[cur][j][0][lane * 16]);
      v16bf bl = *(const v16bf*)(&Bbuf[cur][j][1][lane * 16]);
      acc[j] = __builtin_amdgcn_wmma_f32_16x16x32_bf16(false, ah, false, bh, (short)0, acc[j], false, false);
      acc[j] = __builtin_amdgcn_wmma_f32_16x16x32_bf16(false, al, false, bh, (short)0, acc[j], false, false);
      acc[j] = __builtin_amdgcn_wmma_f32_16x16x32_bf16(false, ah, false, bl, (short)0, acc[j], false, false);
    }
    __syncthreads();                      // all reads done before buffer reuse
  }

  if (!valid) return;
  // store: internal row m = t*32 + b ; logits layout is (b, t, v); d_out[0] is loss
  const int hi = lane >> 4, col = lane & 15;
#pragma unroll
  for (int j = 0; j < 4; ++j) {
    int n = (ng * 4 + j) * 16 + col;
    float bv = b_out[n];
#pragma unroll
    for (int v = 0; v < 8; ++v) {
      int m = mt * 16 + 8 * hi + v;
      int tt = m >> 5, bb = m & 31;
      out[1 + ((size_t)bb * cT + tt) * cV + n] = acc[j][v] + bv;
    }
  }
}

// ---------------- masked NLL: per-row log-softmax ----------------
__global__ void k_loss_rows(const float* __restrict__ out, const int* __restrict__ sent,
                            float* __restrict__ nll, float* __restrict__ cnt) {
  __shared__ float sd[256];
  int r = blockIdx.x;                         // r = b*127 + t
  const float* row = out + 1 + (size_t)r * cV;
  float mx = -INFINITY;
  for (int v = threadIdx.x; v < cV; v += 256) mx = fmaxf(mx, row[v]);
  sd[threadIdx.x] = mx; __syncthreads();
  for (int s = 128; s > 0; s >>= 1) {
    if ((int)threadIdx.x < s) sd[threadIdx.x] = fmaxf(sd[threadIdx.x], sd[threadIdx.x + s]);
    __syncthreads();
  }
  mx = sd[0]; __syncthreads();
  float sum = 0.0f;
  for (int v = threadIdx.x; v < cV; v += 256) sum += __expf(row[v] - mx);
  sd[threadIdx.x] = sum; __syncthreads();
  for (int s = 128; s > 0; s >>= 1) {
    if ((int)threadIdx.x < s) sd[threadIdx.x] += sd[threadIdx.x + s];
    __syncthreads();
  }
  if (threadIdx.x == 0) {
    int bb = r / cT, tt = r % cT;
    int label = sent[bb * 128 + tt + 1];      // sent[:, 1:]
    float v = logf(sd[0]) + mx - row[label];
    int mask = (label != 0);
    nll[r] = mask ? v : 0.0f;
    cnt[r] = mask ? 1.0f : 0.0f;
  }
}

__global__ void k_loss_final(const float* __restrict__ nll, const float* __restrict__ cnt,
                             float* __restrict__ out0) {
  __shared__ float s1[256], s2[256];
  float a = 0.0f, b = 0.0f;
  for (int i = threadIdx.x; i < cM; i += 256) { a += nll[i]; b += cnt[i]; }
  s1[threadIdx.x] = a; s2[threadIdx.x] = b; __syncthreads();
  for (int s = 128; s > 0; s >>= 1) {
    if ((int)threadIdx.x < s) { s1[threadIdx.x] += s1[threadIdx.x + s]; s2[threadIdx.x] += s2[threadIdx.x + s]; }
    __syncthreads();
  }
  if (threadIdx.x == 0) out0[0] = s1[0] / fmaxf(s2[0], 1.0f);
}

// ---------------- host launch ----------------
extern "C" void kernel_launch(void* const* d_in, const int* in_sizes, int n_in,
                              void* d_out, int out_size, void* d_ws, size_t ws_size,
                              hipStream_t stream) {
  (void)in_sizes; (void)n_in; (void)out_size; (void)ws_size;
  const int*   sent    = (const int*)d_in[0];
  const float* wordvec = (const float*)d_in[1];
  const float* W_ih0   = (const float*)d_in[2];
  const float* W_hh0   = (const float*)d_in[3];
  const float* b_ih0   = (const float*)d_in[4];
  const float* b_hh0   = (const float*)d_in[5];
  const float* W_ih1   = (const float*)d_in[6];
  const float* W_hh1   = (const float*)d_in[7];
  const float* b_ih1   = (const float*)d_in[8];
  const float* b_hh1   = (const float*)d_in[9];
  const float* W_out   = (const float*)d_in[10];
  const float* b_out   = (const float*)d_in[11];
  float* out = (float*)d_out;

  // workspace carve-up (~262 MB total)
  char* p = (char*)d_ws;
  auto alloc = [&](size_t bytes) -> char* {
    char* r = p; p += (bytes + 255) & ~(size_t)255; return r;
  };
  unsigned* bar    = (unsigned*)alloc(256);
  float* nll       = (float*)alloc((size_t)cM * 4);
  float* cnt       = (float*)alloc((size_t)cM * 4);
  float* h0        = (float*)alloc((size_t)cB * cH * 4);
  float* h1        = (float*)alloc((size_t)cB * cH * 4);
  float* gh0       = (float*)alloc((size_t)cB * cG * 4);
  float* gi1       = (float*)alloc((size_t)cB * cG * 4);
  float* gh1       = (float*)alloc((size_t)cB * cG * 4);
  float* X         = (float*)alloc((size_t)cM * cKE * 4);
  float* Wt_ih0    = (float*)alloc((size_t)cKE * cG * 4);
  float* Wt_hh0    = (float*)alloc((size_t)cH * cG * 4);
  float* Wt_ih1    = (float*)alloc((size_t)cH * cG * 4);
  float* Wt_hh1    = (float*)alloc((size_t)cH * cG * 4);
  float* Gi0       = (float*)alloc((size_t)cM * cG * 4);
  float* Hs        = (float*)alloc((size_t)cM * cH * 4);
  __bf16* Ahi      = (__bf16*)alloc(((size_t)cMT << 14) * 2);
  __bf16* Alo      = (__bf16*)alloc(((size_t)cMT << 14) * 2);
  __bf16* Bhi      = (__bf16*)alloc(((size_t)cNTO << 14) * 2);
  __bf16* Blo      = (__bf16*)alloc(((size_t)cNTO << 14) * 2);

  const int TB = 256;
  auto nblk = [](size_t n, int tb) { return (unsigned)((n + tb - 1) / tb); };

  k_init<<<nblk((size_t)cB * cH, TB), TB, 0, stream>>>(bar, nll, cnt, h0, h1);
  k_embed<<<nblk((size_t)cM * cKE, TB), TB, 0, stream>>>(sent, wordvec, X);
  k_packT<<<nblk((size_t)cKE * cG, TB), TB, 0, stream>>>(W_ih0, Wt_ih0, cG, cE, cKE);
  k_packT<<<nblk((size_t)cH * cG, TB), TB, 0, stream>>>(W_hh0, Wt_hh0, cG, cH, cH);
  k_packT<<<nblk((size_t)cH * cG, TB), TB, 0, stream>>>(W_ih1, Wt_ih1, cG, cH, cH);
  k_packT<<<nblk((size_t)cH * cG, TB), TB, 0, stream>>>(W_hh1, Wt_hh1, cG, cH, cH);
  k_pack_b<<<nblk((size_t)cNTO << 14, TB), TB, 0, stream>>>(W_out, Bhi, Blo);

  // Gi0 = X @ Wt_ih0 + b_ih0  (4064 x 3072, K=304)
  {
    size_t waves = (size_t)(cM / 16) * (cG / 16);
    k_gemm_bias<<<nblk(waves * 32, TB), TB, 0, stream>>>(X, cKE, Wt_ih0, cG, b_ih0,
                                                         Gi0, cG, cM, cG, cKE);
  }

  // sequential 2-layer GRU scan (persistent, grid-synced)
  k_scan<<<64, TB, 0, stream>>>(Gi0, Wt_hh0, Wt_ih1, Wt_hh1, b_hh0, b_ih1, b_hh1,
                                h0, h1, gh0, gi1, gh1, Hs, bar);

  // pack Hs and run the big split-bf16 output GEMM (LDS-staged B panels)
  k_pack_a<<<nblk((size_t)cMT << 14, TB), TB, 0, stream>>>(Hs, Ahi, Alo);
  k_gemm_out<<<cMG * cNGO, TB, 0, stream>>>(Ahi, Alo, Bhi, Blo, b_out, out);

  // loss
  k_loss_rows<<<cM, TB, 0, stream>>>(out, sent, nll, cnt);
  k_loss_final<<<1, TB, 0, stream>>>(nll, cnt, out);
}
